// RandomSSM_63385127354473
// MI455X (gfx1250) — matrix operand order repaired and checked by reading, not compiled, not run on hardware
//
#include <hip/hip_runtime.h>

// ---------------------------------------------------------------------------
// RandomSSM on gfx1250:
//   Bu = u @ B^T            (bf16 WMMA, f32 acc)  -> d_out used as f32 scratch
//   X  = scan(diag(A), Bu)  (fp32 serial in T, parallel over batch*n) -> bf16
//   y  = X @ C^T + u @ D^T  (bf16 WMMA, f32 acc, K=1024+1024) -> d_out
// Wave tile 64x64 (16 WMMA : 16 b128 loads per K=32 step), block tile 128x256.
// ---------------------------------------------------------------------------

typedef __attribute__((ext_vector_type(16))) __bf16 v16bf;
typedef __attribute__((ext_vector_type(8)))  __bf16 v8bf;
typedef __attribute__((ext_vector_type(8)))  float  v8f;

#define BATCH   8
#define TSEQ    2048
#define NDIM    1024
#define MROWS   (BATCH * TSEQ)   // 16384

union FragBF {
    v16bf v;
    v8bf  h[2];
};

// ---------------- fp32 -> bf16 conversion (RNE via native cast) -------------
__global__ void f32_to_bf16_kernel(const float* __restrict__ in,
                                   __bf16* __restrict__ out, int n) {
    int i = blockIdx.x * blockDim.x + threadIdx.x;
    if (i < n) out[i] = (__bf16)in[i];
}

// ---------------- WMMA fragment loaders ------------------------------------
// A matrix 16x32 bf16 (ISA 7.12.2): lane l holds row l&15.
//   elems 0..7  -> K = hi8 + j        (hi8 = 8 for lanes 16..31)
//   elems 8..15 -> K = 16 + hi8 + j-8
__device__ __forceinline__ v16bf load_frag_a(const __bf16* __restrict__ tile, int ld) {
    const int lane = threadIdx.x & 31;
    const int r    = lane & 15;
    const int hi8  = (lane >> 4) << 3;           // 0 or 8
    const __bf16* p = tile + (size_t)r * ld + hi8;
    FragBF f;
    f.h[0] = *(const v8bf*)(p);                  // K = hi8 .. hi8+7
    f.h[1] = *(const v8bf*)(p + 16);             // K = 16+hi8 .. 16+hi8+7
    return f.v;
}

// B matrix 32x16 bf16: lane l holds column l&15, K same order as A (hi8+j, 16+hi8+j-8).
// Weight is stored row-major W[n][k] (k fastest), i.e. columns of B are rows of W.
__device__ __forceinline__ v16bf load_frag_b(const __bf16* __restrict__ tile, int ld) {
    const int lane = threadIdx.x & 31;
    const int n    = lane & 15;
    const int hi16 = (lane >> 4) << 3;           // 0 or 8
    const __bf16* p = tile + (size_t)n * ld + hi16;
    FragBF f;
    f.h[0] = *(const v8bf*)(p);                  // K = hi16 .. hi16+7
    f.h[1] = *(const v8bf*)(p + 16);             // K = 16+hi16 .. 16+hi16+7
    return f.v;
}

// ---------------- one K-segment of the 64x64-per-wave GEMM ------------------
__device__ __forceinline__ void mma_segment(v8f (&acc)[4][4],
                                            const __bf16* __restrict__ act,
                                            const __bf16* __restrict__ w,
                                            int m0, int n0, int K) {
    for (int k = 0; k < K; k += 32) {
        v16bf a[4], b[4];
#pragma unroll
        for (int i = 0; i < 4; ++i)
            a[i] = load_frag_a(act + (size_t)(m0 + 16 * i) * K + k, K);
#pragma unroll
        for (int j = 0; j < 4; ++j)
            b[j] = load_frag_b(w + (size_t)(n0 + 16 * j) * K + k, K);

#pragma unroll
        for (int i = 0; i < 4; ++i)
#pragma unroll
            for (int j = 0; j < 4; ++j)
                acc[i][j] = __builtin_amdgcn_wmma_f32_16x16x32_bf16(
                    false, a[i], false, b[j], (short)0, acc[i][j], false, false);
    }
}

// ---------------- GEMM: out[M][N] = sum_seg act_s[M][K] @ w_s[N][K]^T -------
// Block = 256 threads = 8 waves; wave tile 64(M) x 64(N); block tile 128x256.
template <int NSEG>
__global__ __launch_bounds__(256)
void gemm_bf16_kernel(float* __restrict__ out,
                      const __bf16* __restrict__ act0, const __bf16* __restrict__ w0,
                      const __bf16* __restrict__ act1, const __bf16* __restrict__ w1,
                      int K, int N) {
    const int wave = threadIdx.x >> 5;
    const int wm   = wave & 1;           // 2 waves along M  -> 128
    const int wn   = wave >> 1;          // 4 waves along N  -> 256
    const int m0   = blockIdx.x * 128 + wm * 64;
    const int n0   = blockIdx.y * 256 + wn * 64;

    v8f acc[4][4];
#pragma unroll
    for (int i = 0; i < 4; ++i)
#pragma unroll
        for (int j = 0; j < 4; ++j)
            acc[i][j] = (v8f){0.f, 0.f, 0.f, 0.f, 0.f, 0.f, 0.f, 0.f};

    mma_segment(acc, act0, w0, m0, n0, K);
    if constexpr (NSEG == 2)
        mma_segment(acc, act1, w1, m0, n0, K);

    // C/D layout: VGPR v, lane l -> row ((l>=16)?8:0)+v, col l&15
    const int lane = threadIdx.x & 31;
    const int col  = lane & 15;
    const int rb   = (lane >> 4) << 3;
#pragma unroll
    for (int i = 0; i < 4; ++i)
#pragma unroll
        for (int j = 0; j < 4; ++j) {
            float* op = out + (size_t)(m0 + 16 * i + rb) * N + (n0 + 16 * j + col);
#pragma unroll
            for (int v = 0; v < 8; ++v)
                op[(size_t)v * N] = acc[i][j][v];
        }
}

// ---------------- SSM scan: x_t = a*x_{t-1} + Bu_t, emit bf16 X -------------
// One thread per (batch, channel); lanes cover consecutive channels so every
// time-step f32 read is a fully coalesced 128B burst per wave (bf16 write: 64B).
__global__ __launch_bounds__(256)
void ssm_scan_kernel(const float* __restrict__ Bu,   // [MROWS][NDIM] f32
                     const float* __restrict__ A,    // [NDIM][NDIM] (diag used)
                     __bf16* __restrict__ Xb) {      // [MROWS][NDIM] bf16
    const int idx = blockIdx.x * blockDim.x + threadIdx.x;  // 0 .. BATCH*NDIM-1
    const int b   = idx / NDIM;
    const int ch  = idx - b * NDIM;
    const float a = A[(size_t)ch * NDIM + ch];

    const float* __restrict__ bu = Bu + (size_t)b * TSEQ * NDIM + ch;
    __bf16* __restrict__ xo      = Xb + (size_t)b * TSEQ * NDIM + ch;

    float x = 0.f;
    for (int t = 0; t < TSEQ; ++t) {
        x = fmaf(x, a, bu[(size_t)t * NDIM]);
        xo[(size_t)t * NDIM] = (__bf16)x;
    }
}

// ---------------------------------------------------------------------------
extern "C" void kernel_launch(void* const* d_in, const int* in_sizes, int n_in,
                              void* d_out, int out_size, void* d_ws, size_t ws_size,
                              hipStream_t stream) {
    const float* u = (const float*)d_in[0];   // [8][2048][1024]
    const float* A = (const float*)d_in[1];   // [1024][1024]
    const float* B = (const float*)d_in[2];   // [1024][1024]
    const float* C = (const float*)d_in[3];   // [1024][1024]
    const float* D = (const float*)d_in[4];   // [1024][1024]
    float* out = (float*)d_out;               // [8][2048][1024] f32; reused as Bu scratch

    const size_t U_ELEMS = (size_t)MROWS * NDIM;      // 16,777,216
    const size_t W_ELEMS = (size_t)NDIM * NDIM;       //  1,048,576

    char* ws = (char*)d_ws;
    __bf16* u_bf = (__bf16*)(ws);                                   // 32 MiB
    __bf16* B_bf = (__bf16*)(ws + U_ELEMS * 2);                     //  2 MiB
    __bf16* C_bf = (__bf16*)(ws + U_ELEMS * 2 + W_ELEMS * 2);       //  2 MiB
    __bf16* D_bf = (__bf16*)(ws + U_ELEMS * 2 + W_ELEMS * 4);       //  2 MiB
    __bf16* X_bf = (__bf16*)(ws + U_ELEMS * 2 + W_ELEMS * 6);       // 32 MiB

    // 1) fp32 -> bf16 operand conversions
    f32_to_bf16_kernel<<<(int)(U_ELEMS / 256), 256, 0, stream>>>(u, u_bf, (int)U_ELEMS);
    f32_to_bf16_kernel<<<(int)(W_ELEMS / 256), 256, 0, stream>>>(B, B_bf, (int)W_ELEMS);
    f32_to_bf16_kernel<<<(int)(W_ELEMS / 256), 256, 0, stream>>>(C, C_bf, (int)W_ELEMS);
    f32_to_bf16_kernel<<<(int)(W_ELEMS / 256), 256, 0, stream>>>(D, D_bf, (int)W_ELEMS);

    // 2) Bu = u @ B^T  (f32 result staged in d_out)
    dim3 ggrid(MROWS / 128, NDIM / 256);   // 128 x 4
    gemm_bf16_kernel<1><<<ggrid, 256, 0, stream>>>(out, u_bf, B_bf,
                                                   (const __bf16*)nullptr,
                                                   (const __bf16*)nullptr,
                                                   NDIM, NDIM);

    // 3) serial-in-time scan, parallel over batch*channels; writes bf16 X
    ssm_scan_kernel<<<(BATCH * NDIM) / 256, 256, 0, stream>>>(out, A, X_bf);

    // 4) y = X @ C^T + u @ D^T  (two K=1024 segments, overwrites d_out)
    gemm_bf16_kernel<2><<<ggrid, 256, 0, stream>>>(out, X_bf, C_bf, u_bf, D_bf,
                                                   NDIM, NDIM);
}
